// SALA_6382321402435
// MI455X (gfx1250) — matrix operand order repaired and checked by reading, not compiled, not run on hardware
//
#include <hip/hip_runtime.h>
#include <hip/hip_bf16.h>
#include <math.h>

typedef _Float16 half_t;
typedef __attribute__((ext_vector_type(16))) _Float16 v16h;
typedef __attribute__((ext_vector_type(8)))  _Float16 v8h;
typedef __attribute__((ext_vector_type(8)))  float    v8f;

#define DMODEL 512
#define NTOK   50176   // 4*112*112
#define HWDIM  112
#define NSUB   784     // 28*28
#define MSUB   3136    // 4*784

// ---------------------------------------------------------------- WMMA helpers
__device__ __forceinline__ v8f wmma16(v16h a, v16h b, v8f c) {
  return __builtin_amdgcn_wmma_f32_16x16x32_f16(false, a, false, b, (short)0, c,
                                                false, false);
}

// A fragment (16x32, f16): lane<16 -> row=lane, K {0..7,16..23};
// lane>=16 -> row=lane-16, K {8..15,24..31}   (ISA 7.12.2 layout)
__device__ __forceinline__ v16h frag_a(const half_t* base, int stride, int lane) {
  int m  = lane & 15;
  int kb = (lane < 16) ? 0 : 8;
  const half_t* p = base + m * stride + kb;
  v8h lo = *(const v8h*)(p);
  v8h hi = *(const v8h*)(p + 16);
  return __builtin_shufflevector(lo, hi, 0,1,2,3,4,5,6,7,8,9,10,11,12,13,14,15);
}

// B fragment (32x16, f16) sourced from LDS tile stored [n][k]:
// lane<16 -> col=lane, K {0..7,16..23} ; lane>=16 -> col=lane-16, K {8..15,24..31}
__device__ __forceinline__ v16h frag_b(const half_t* base, int stride, int lane) {
  int n  = lane & 15;
  int kb = (lane < 16) ? 0 : 8;
  const half_t* p = base + n * stride + kb;
  v8h lo = *(const v8h*)(p);
  v8h hi = *(const v8h*)(p + 16);
  return __builtin_shufflevector(lo, hi, 0,1,2,3,4,5,6,7,8,9,10,11,12,13,14,15);
}

// ---------------------------------------------------------------- generic GEMM
// C[M,N] = scale * A[M,K] x B + bias.  B is (K,N) row-major, or (N,K) if TRANSB.
// Block tile 128x128, K-step 64 (2 WMMA K-slices), 8 waves (4x2), wave 32x64.
template<typename AT, typename BT, typename CT, bool TRANSB>
__global__ __launch_bounds__(256) void gemm_wmma(
    const AT* __restrict__ A, const BT* __restrict__ Bm, CT* __restrict__ C,
    const float* __restrict__ bias,
    int M, int N, int K, int lda, int ldb, int ldc,
    long sA, long sB, long sC, float scale)
{
  __shared__ __align__(16) char smem[2 * 128 * 72 * sizeof(half_t)];  // 36 KB
  half_t (*As)[72] = (half_t(*)[72])smem;
  half_t (*Bs)[72] = (half_t(*)[72])(smem + 128 * 72 * sizeof(half_t));

  const int tid  = threadIdx.x;
  const int lane = tid & 31;
  const int wave = tid >> 5;
  const int wm   = (wave >> 1) * 32;
  const int wn   = (wave & 1) * 64;
  const int m0   = blockIdx.x * 128;
  const int n0   = blockIdx.y * 128;
  A  += (long)blockIdx.z * sA;
  Bm += (long)blockIdx.z * sB;
  C  += (long)blockIdx.z * sC;

  v8f acc[2][4];
#pragma unroll
  for (int i = 0; i < 2; ++i)
#pragma unroll
    for (int j = 0; j < 4; ++j) acc[i][j] = {};

  for (int kk = 0; kk < K; kk += 64) {
    // prefetch next K-panel (global_prefetch_b8)
    if (kk + 64 < K) {
      int prow = tid >> 4;
      int pkq  = (tid & 15) << 2;
      if (m0 + prow < M)
        __builtin_prefetch(A + (long)(m0 + prow) * lda + kk + 64 + pkq, 0, 0);
      if constexpr (TRANSB) {
        if (n0 + prow < N)
          __builtin_prefetch(Bm + (long)(n0 + prow) * ldb + kk + 64 + pkq, 0, 0);
      }
    }
    // ---- A tile (128x64) -> As[row][k]
#pragma unroll
    for (int i = 0; i < 8; ++i) {
      int idx = tid + (i << 8);        // 0..2047 slots of 4
      int row = idx >> 4;
      int kq  = (idx & 15) << 2;
      int gr = m0 + row, gk = kk + kq;
      half_t h[4] = {(half_t)0.f, (half_t)0.f, (half_t)0.f, (half_t)0.f};
      if (gr < M) {
        const AT* ap = A + (long)gr * lda + gk;
        if (gk + 3 < K) {
#pragma unroll
          for (int e = 0; e < 4; ++e) h[e] = (half_t)(float)ap[e];
        } else {
#pragma unroll
          for (int e = 0; e < 4; ++e) if (gk + e < K) h[e] = (half_t)(float)ap[e];
        }
      }
#pragma unroll
      for (int e = 0; e < 4; ++e) As[row][kq + e] = h[e];
    }
    // ---- B tile -> Bs[n][k]
    if constexpr (TRANSB) {            // B is (N,K) row-major: plain row copy
#pragma unroll
      for (int i = 0; i < 8; ++i) {
        int idx = tid + (i << 8);
        int row = idx >> 4;
        int kq  = (idx & 15) << 2;
        int gn = n0 + row, gk = kk + kq;
        half_t h[4] = {(half_t)0.f, (half_t)0.f, (half_t)0.f, (half_t)0.f};
        if (gn < N) {
          const BT* bp = Bm + (long)gn * ldb + gk;
          if (gk + 3 < K) {
#pragma unroll
            for (int e = 0; e < 4; ++e) h[e] = (half_t)(float)bp[e];
          } else {
#pragma unroll
            for (int e = 0; e < 4; ++e) if (gk + e < K) h[e] = (half_t)(float)bp[e];
          }
        }
#pragma unroll
        for (int e = 0; e < 4; ++e) Bs[row][kq + e] = h[e];
      }
    } else {                           // B is (K,N) row-major: transpose in
#pragma unroll
      for (int i = 0; i < 8; ++i) {
        int idx = tid + (i << 8);
        int kr = idx >> 5;             // 0..63
        int nq = (idx & 31) << 2;
        int gk = kk + kr;
#pragma unroll
        for (int e = 0; e < 4; ++e) {
          int gn = n0 + nq + e;
          float v = 0.f;
          if (gk < K && gn < N) v = (float)Bm[(long)gk * ldb + gn];
          Bs[nq + e][kr] = (half_t)v;
        }
      }
    }
    __syncthreads();
#pragma unroll
    for (int ks2 = 0; ks2 < 2; ++ks2) {
      int ko = ks2 << 5;
      v16h a0 = frag_a(&As[wm][ko],      72, lane);
      v16h a1 = frag_a(&As[wm + 16][ko], 72, lane);
      v16h b0 = frag_b(&Bs[wn][ko],      72, lane);
      v16h b1 = frag_b(&Bs[wn + 16][ko], 72, lane);
      v16h b2 = frag_b(&Bs[wn + 32][ko], 72, lane);
      v16h b3 = frag_b(&Bs[wn + 48][ko], 72, lane);
      acc[0][0] = wmma16(a0, b0, acc[0][0]);
      acc[0][1] = wmma16(a0, b1, acc[0][1]);
      acc[0][2] = wmma16(a0, b2, acc[0][2]);
      acc[0][3] = wmma16(a0, b3, acc[0][3]);
      acc[1][0] = wmma16(a1, b0, acc[1][0]);
      acc[1][1] = wmma16(a1, b1, acc[1][1]);
      acc[1][2] = wmma16(a1, b2, acc[1][2]);
      acc[1][3] = wmma16(a1, b3, acc[1][3]);
    }
    __syncthreads();
  }

  const int lm = (lane >> 4) << 3;     // M offset inside 16x16 tile (0 or 8)
  const int ln = lane & 15;
  if constexpr (sizeof(CT) == 2) {
    if (m0 + 128 <= M && n0 + 128 <= N) {
      // stage wave tile in (reused) LDS, then 16B coalesced stores
      half_t* Cw = (half_t*)smem + wave * (32 * 72);
#pragma unroll
      for (int mt = 0; mt < 2; ++mt)
#pragma unroll
        for (int nt = 0; nt < 4; ++nt) {
          int gn = n0 + wn + nt * 16 + ln;
          float bv = (bias != nullptr) ? bias[gn] : 0.f;
#pragma unroll
          for (int r = 0; r < 8; ++r)
            Cw[(mt * 16 + lm + r) * 72 + nt * 16 + ln] =
                (half_t)(acc[mt][nt][r] * scale + bv);
        }
      // LDS is in-order within a wave; each wave reads only its own region
      long gm0 = m0 + wm, gn0 = n0 + wn;
#pragma unroll
      for (int it = 0; it < 8; ++it) {
        int idx = it * 32 + lane;
        int row = idx >> 3;
        int seg = idx & 7;
        uint4 v = *(const uint4*)(Cw + row * 72 + seg * 8);
        *(uint4*)((half_t*)C + (gm0 + row) * ldc + gn0 + seg * 8) = v;
      }
      return;
    }
  }
  // fallback: guarded elementwise stores (partial tiles / f32 output)
#pragma unroll
  for (int mt = 0; mt < 2; ++mt)
#pragma unroll
    for (int nt = 0; nt < 4; ++nt) {
      int gn = n0 + wn + nt * 16 + ln;
      float bv = (bias != nullptr && gn < N) ? bias[gn] : 0.f;
#pragma unroll
      for (int r = 0; r < 8; ++r) {
        int gm = m0 + wm + mt * 16 + lm + r;
        if (gm < M && gn < N) {
          float v = acc[mt][nt][r] * scale + bv;
          C[(long)gm * ldc + gn] = (CT)v;
        }
      }
    }
}

// ----------------------------------------------------- fused window attention
// One wave per (window, head).  L=49 padded to 64, HD=32.
__global__ __launch_bounds__(32) void win_attn(
    const half_t* __restrict__ Q, const half_t* __restrict__ K,
    const half_t* __restrict__ V, half_t* __restrict__ AO)
{
  __shared__ half_t qs[64][40];
  __shared__ half_t ks[64][40];
  __shared__ half_t vT[32][72];   // vT[dim][token]
  __shared__ float  Sb[64][65];
  __shared__ half_t Pb[64][72];

  const int lane = threadIdx.x;
  const int wid  = blockIdx.x;        // 0..1023
  const int h    = blockIdx.y;        // 0..15
  const int b    = wid >> 8;
  const int rem  = wid & 255;
  const int wh   = rem >> 4;
  const int ww   = rem & 15;

  for (int row = lane; row < 64; row += 32) {
    if (row < 49) {
      int i = row / 7, j = row % 7;
      int n = (wh * 7 + i) * HWDIM + (ww * 7 + j);
      long grow = ((long)(b * 12544 + n)) * DMODEL + h * 32;
      const v8h* qp = (const v8h*)(Q + grow);
      const v8h* kp = (const v8h*)(K + grow);
#pragma unroll
      for (int t = 0; t < 4; ++t) {
        *(v8h*)&qs[row][t * 8] = qp[t];
        *(v8h*)&ks[row][t * 8] = kp[t];
      }
      const half_t* vp = V + grow;
#pragma unroll
      for (int d = 0; d < 32; ++d) vT[d][row] = vp[d];
    } else {
      v8h z = {};
#pragma unroll
      for (int t = 0; t < 4; ++t) {
        *(v8h*)&qs[row][t * 8] = z;
        *(v8h*)&ks[row][t * 8] = z;
      }
#pragma unroll
      for (int d = 0; d < 32; ++d) vT[d][row] = (half_t)0.f;
    }
  }
  __syncthreads();

  // S = q k^T / sqrt(32)
  const float sscale = 0.1767766952966369f;
  const int lm = (lane >> 4) << 3;
  const int ln = lane & 15;
#pragma unroll
  for (int mt = 0; mt < 4; ++mt) {
    v16h a = frag_a(&qs[mt * 16][0], 40, lane);
#pragma unroll
    for (int nt = 0; nt < 4; ++nt) {
      v16h bf = frag_b(&ks[nt * 16][0], 40, lane);
      v8f c = {};
      c = wmma16(a, bf, c);
      int mb = mt * 16 + lm;
      int nc = nt * 16 + ln;
#pragma unroll
      for (int r = 0; r < 8; ++r) Sb[mb + r][nc] = c[r] * sscale;
    }
  }
  __syncthreads();

  // softmax over 49 valid columns; emit P (f16, zero-padded)
  for (int row = lane; row < 64; row += 32) {
    if (row < 49) {
      float mx = -3.0e38f;
      for (int c = 0; c < 49; ++c) mx = fmaxf(mx, Sb[row][c]);
      float sum = 0.f;
      for (int c = 0; c < 49; ++c) {
        float e = __expf(Sb[row][c] - mx);
        Sb[row][c] = e;
        sum += e;
      }
      float inv = 1.0f / sum;
      for (int c = 0; c < 64; ++c)
        Pb[row][c] = (c < 49) ? (half_t)(Sb[row][c] * inv) : (half_t)0.f;
    } else {
      for (int c = 0; c < 64; ++c) Pb[row][c] = (half_t)0.f;
    }
  }
  __syncthreads();

  // O = P @ V : M=64, N=32, K=64
  v8f o[4][2];
#pragma unroll
  for (int mt = 0; mt < 4; ++mt)
#pragma unroll
    for (int nt = 0; nt < 2; ++nt) o[mt][nt] = {};
#pragma unroll
  for (int kt = 0; kt < 2; ++kt) {
#pragma unroll
    for (int mt = 0; mt < 4; ++mt) {
      v16h a = frag_a(&Pb[mt * 16][kt * 32], 72, lane);
#pragma unroll
      for (int nt = 0; nt < 2; ++nt) {
        v16h bf = frag_b(&vT[nt * 16][kt * 32], 72, lane);
        o[mt][nt] = wmma16(a, bf, o[mt][nt]);
      }
    }
  }

#pragma unroll
  for (int mt = 0; mt < 4; ++mt)
#pragma unroll
    for (int nt = 0; nt < 2; ++nt) {
      int mb = mt * 16 + lm;
      int nc = h * 32 + nt * 16 + ln;
#pragma unroll
      for (int r = 0; r < 8; ++r) {
        int m = mb + r;
        if (m < 49) {
          int i = m / 7, j = m % 7;
          long grow = ((long)(b * 12544 + (wh * 7 + i) * HWDIM + ww * 7 + j)) * DMODEL;
          AO[grow + nc] = (half_t)o[mt][nt][r];
        }
      }
    }
}

// --------------------------------------------------------------- row softmax
__global__ __launch_bounds__(256) void softmax_rows(float* __restrict__ S, int cols)
{
  long row = blockIdx.x;
  float* p = S + row * cols;
  __shared__ float red[256];
  int tid = threadIdx.x;
  float mx = -3.0e38f;
  for (int c = tid; c < cols; c += 256) mx = fmaxf(mx, p[c]);
  red[tid] = mx;
  __syncthreads();
  for (int s = 128; s > 0; s >>= 1) {
    if (tid < s) red[tid] = fmaxf(red[tid], red[tid + s]);
    __syncthreads();
  }
  mx = red[0];
  __syncthreads();
  float sum = 0.f;
  for (int c = tid; c < cols; c += 256) {
    float e = __expf(p[c] - mx);
    p[c] = e;
    sum += e;
  }
  red[tid] = sum;
  __syncthreads();
  for (int s = 128; s > 0; s >>= 1) {
    if (tid < s) red[tid] += red[tid + s];
    __syncthreads();
  }
  float inv = 1.0f / red[0];
  for (int c = tid; c < cols; c += 256) p[c] *= inv;
}

// ------------------------- fold 2D DCT into Wsp, emit transposed f16 (N,K)
__global__ __launch_bounds__(256) void build_w2t(const float* __restrict__ Wsp,
                                                 half_t* __restrict__ W2t)
{
  __shared__ float Bas[8][8];
  int tid = threadIdx.x;
  if (tid < 64) {
    int kk = tid >> 3, ii = tid & 7;
    float v = __cosf(3.14159265358979323846f * (ii + 0.5f) * kk / 8.0f) * 0.5f;
    if (kk == 0) v *= 0.70710678118654752f;
    Bas[kk][ii] = v;          // Bas[freq][spatial]
  }
  __syncthreads();
  long gid = (long)blockIdx.x * 256 + tid;   // 512*192 total
  if (gid >= 512 * 192) return;
  int d  = (int)(gid / 192);
  int rv = (int)(gid % 192);
  int c = rv >> 6, j = (rv >> 3) & 7, l = rv & 7;
  float s = 0.f;
#pragma unroll
  for (int i = 0; i < 8; ++i)
#pragma unroll
    for (int k = 0; k < 8; ++k)
      s += Bas[i][j] * Bas[k][l] * Wsp[(c * 64 + i * 8 + k) * DMODEL + d];
  W2t[(long)d * 192 + rv] = (half_t)s;
}

// ------------------------------------- gather ::4 subsampled patch vectors
__global__ __launch_bounds__(256) void patch_gather(const float* __restrict__ P,
                                                    half_t* __restrict__ PS)
{
  long gid = (long)blockIdx.x * 256 + threadIdx.x;   // 3136*192
  if (gid >= (long)MSUB * 192) return;
  int row = (int)(gid / 192);
  int c   = (int)(gid % 192);
  int b = row / NSUB, s = row % NSUB;
  int si = s / 28, sj = s % 28;
  int n = (si * 4) * HWDIM + sj * 4;
  PS[(long)row * 192 + c] = (half_t)P[((long)b * 192 + c) * 12544 + n];
}

// -------------------------------------------- tiled transpose f32 -> f16
// W is (R x C) row-major; Wt is (C x R) row-major f16.
__global__ __launch_bounds__(256) void transpose_to_f16(const float* __restrict__ W,
                                                        half_t* __restrict__ Wt,
                                                        int R, int Ccols)
{
  __shared__ float tile[32][33];
  int bx = blockIdx.x * 32;   // column base
  int by = blockIdx.y * 32;   // row base
  int tx = threadIdx.x & 31, ty = threadIdx.x >> 5;
  for (int r = ty; r < 32; r += 8) {
    int gr = by + r, gc = bx + tx;
    tile[r][tx] = (gr < R && gc < Ccols) ? W[(long)gr * Ccols + gc] : 0.f;
  }
  __syncthreads();
  for (int r = ty; r < 32; r += 8) {
    int gc = bx + r, gr = by + tx;
    if (gc < Ccols && gr < R) Wt[(long)gc * R + gr] = (half_t)tile[tx][r];
  }
}

// ------------------------------------------------------- f32 -> f16 convert
__global__ __launch_bounds__(256) void f32_to_f16(const float* __restrict__ in,
                                                  half_t* __restrict__ outp, long n)
{
  long g = ((long)blockIdx.x * 256 + threadIdx.x) * 4;
  if (g + 3 < n) {
    float4 v = *(const float4*)(in + g);
    outp[g + 0] = (half_t)v.x;
    outp[g + 1] = (half_t)v.y;
    outp[g + 2] = (half_t)v.z;
    outp[g + 3] = (half_t)v.w;
  } else {
    for (int e = 0; e < 4; ++e)
      if (g + e < n) outp[g + e] = (half_t)in[g + e];
  }
}

// -------------------------------------------------- combine + LayerNorm
__global__ __launch_bounds__(128) void combine_ln(
    const float* __restrict__ tokens, const half_t* __restrict__ SP,
    const half_t* __restrict__ SO, const float* __restrict__ alpha_p,
    float* __restrict__ out)
{
  int t = blockIdx.x;                 // 0..50175
  int b = t / 12544, n = t % 12544;
  int r = n / HWDIM, c = n % HWDIM;
  long base  = (long)t * DMODEL;
  long sbase = ((long)(b * NSUB + (r >> 2) * 28 + (c >> 2))) * DMODEL;
  float al = 1.f / (1.f + __expf(-alpha_p[0]));
  float om = 1.f - al;
  int tid = threadIdx.x;
  float v[4];
  float lsum = 0.f, lsq = 0.f;
#pragma unroll
  for (int e = 0; e < 4; ++e) {
    int d = tid + e * 128;
    float x = tokens[base + d] + al * (float)SP[base + d] + om * (float)SO[sbase + d];
    v[e] = x;
    lsum += x;
    lsq  += x * x;
  }
  __shared__ float rs[128], rq[128];
  rs[tid] = lsum; rq[tid] = lsq;
  __syncthreads();
  for (int s = 64; s > 0; s >>= 1) {
    if (tid < s) { rs[tid] += rs[tid + s]; rq[tid] += rq[tid + s]; }
    __syncthreads();
  }
  float mean = rs[0] * (1.f / 512.f);
  float var  = rq[0] * (1.f / 512.f) - mean * mean;
  float inv  = rsqrtf(var + 1e-5f);
#pragma unroll
  for (int e = 0; e < 4; ++e) {
    int d = tid + e * 128;
    out[base + d] = (v[e] - mean) * inv;
  }
}

// ------------------------------------------------------------------- launch
extern "C" void kernel_launch(void* const* d_in, const int* in_sizes, int n_in,
                              void* d_out, int out_size, void* d_ws, size_t ws_size,
                              hipStream_t stream)
{
  const float* tokens  = (const float*)d_in[0];
  const float* patches = (const float*)d_in[1];
  const float* Wq = (const float*)d_in[2];  const float* bq = (const float*)d_in[3];
  const float* Wk = (const float*)d_in[4];  const float* bk = (const float*)d_in[5];
  const float* Wv = (const float*)d_in[6];  const float* bv = (const float*)d_in[7];
  const float* Wo = (const float*)d_in[8];  const float* bo = (const float*)d_in[9];
  const float* Wsp = (const float*)d_in[10]; const float* bsp = (const float*)d_in[11];
  const float* Wsq = (const float*)d_in[12]; const float* bsq = (const float*)d_in[13];
  const float* Wsk = (const float*)d_in[14]; const float* bsk = (const float*)d_in[15];
  const float* Wsv = (const float*)d_in[16]; const float* bsv = (const float*)d_in[17];
  const float* Wso = (const float*)d_in[18]; const float* bso = (const float*)d_in[19];
  const float* alpha = (const float*)d_in[20];
  float* out = (float*)d_out;

  char* ws = (char*)d_ws;
  auto carve = [&](size_t bytes) {
    char* p = ws;
    ws += (bytes + 255) & ~(size_t)255;
    return p;
  };
  half_t* tok16 = (half_t*)carve((size_t)NTOK * DMODEL * 2);
  half_t* Qw  = (half_t*)carve((size_t)NTOK * DMODEL * 2);
  half_t* Kw  = (half_t*)carve((size_t)NTOK * DMODEL * 2);
  half_t* Vw  = (half_t*)carve((size_t)NTOK * DMODEL * 2);
  half_t* AOw = (half_t*)carve((size_t)NTOK * DMODEL * 2);
  half_t* SPw = (half_t*)carve((size_t)NTOK * DMODEL * 2);
  half_t* Wqt = (half_t*)carve((size_t)DMODEL * DMODEL * 2);
  half_t* Wkt = (half_t*)carve((size_t)DMODEL * DMODEL * 2);
  half_t* Wvt = (half_t*)carve((size_t)DMODEL * DMODEL * 2);
  half_t* Wot = (half_t*)carve((size_t)DMODEL * DMODEL * 2);
  half_t* Wsqt = (half_t*)carve((size_t)DMODEL * DMODEL * 2);
  half_t* Wskt = (half_t*)carve((size_t)DMODEL * DMODEL * 2);
  half_t* Wsvt = (half_t*)carve((size_t)DMODEL * DMODEL * 2);
  half_t* Wsot = (half_t*)carve((size_t)DMODEL * DMODEL * 2);
  half_t* W2t = (half_t*)carve((size_t)DMODEL * 192 * 2);
  half_t* PSw = (half_t*)carve((size_t)MSUB * 192 * 2);
  half_t* SDw = (half_t*)carve((size_t)MSUB * DMODEL * 2);
  half_t* QSw = (half_t*)carve((size_t)MSUB * DMODEL * 2);
  half_t* KSw = (half_t*)carve((size_t)MSUB * DMODEL * 2);
  half_t* VSw = (half_t*)carve((size_t)MSUB * DMODEL * 2);
  float*  SSw = (float*)carve((size_t)4 * NSUB * NSUB * 4);
  half_t* SOP = (half_t*)carve((size_t)MSUB * DMODEL * 2);
  half_t* SOw = (half_t*)carve((size_t)MSUB * DMODEL * 2);

  const dim3 gT(16, 16, 1);
  const dim3 gBig(392, 4, 1);   // M=50176/128, N=512/128
  const dim3 gSub(25, 4, 1);    // ceil(3136/128)

  // ---- one-time prep: f16 conversions / weight transposes
  f32_to_f16<<<((long)NTOK * DMODEL + 1023) / 1024, 256, 0, stream>>>(
      tokens, tok16, (long)NTOK * DMODEL);
  transpose_to_f16<<<gT, 256, 0, stream>>>(Wq,  Wqt,  DMODEL, DMODEL);
  transpose_to_f16<<<gT, 256, 0, stream>>>(Wk,  Wkt,  DMODEL, DMODEL);
  transpose_to_f16<<<gT, 256, 0, stream>>>(Wv,  Wvt,  DMODEL, DMODEL);
  transpose_to_f16<<<gT, 256, 0, stream>>>(Wo,  Wot,  DMODEL, DMODEL);
  transpose_to_f16<<<gT, 256, 0, stream>>>(Wsq, Wsqt, DMODEL, DMODEL);
  transpose_to_f16<<<gT, 256, 0, stream>>>(Wsk, Wskt, DMODEL, DMODEL);
  transpose_to_f16<<<gT, 256, 0, stream>>>(Wsv, Wsvt, DMODEL, DMODEL);
  transpose_to_f16<<<gT, 256, 0, stream>>>(Wso, Wsot, DMODEL, DMODEL);
  build_w2t<<<384, 256, 0, stream>>>(Wsp, W2t);
  patch_gather<<<((long)MSUB * 192 + 255) / 256, 256, 0, stream>>>(patches, PSw);

  // ---- spatial path: Q/K/V projections (all-f16 tiles)
  gemm_wmma<half_t, half_t, half_t, true><<<gBig, 256, 0, stream>>>(
      tok16, Wqt, Qw, bq, NTOK, DMODEL, DMODEL, DMODEL, DMODEL, DMODEL, 0, 0, 0, 1.f);
  gemm_wmma<half_t, half_t, half_t, true><<<gBig, 256, 0, stream>>>(
      tok16, Wkt, Kw, bk, NTOK, DMODEL, DMODEL, DMODEL, DMODEL, DMODEL, 0, 0, 0, 1.f);
  gemm_wmma<half_t, half_t, half_t, true><<<gBig, 256, 0, stream>>>(
      tok16, Wvt, Vw, bv, NTOK, DMODEL, DMODEL, DMODEL, DMODEL, DMODEL, 0, 0, 0, 1.f);

  // ---- fused 7x7 window attention, one wave per (window, head)
  win_attn<<<dim3(1024, 16, 1), 32, 0, stream>>>(Qw, Kw, Vw, AOw);

  // ---- output projection
  gemm_wmma<half_t, half_t, half_t, true><<<gBig, 256, 0, stream>>>(
      AOw, Wot, SPw, bo, NTOK, DMODEL, DMODEL, DMODEL, DMODEL, DMODEL, 0, 0, 0, 1.f);

  // ---- spectral descriptor projection (DCT folded into W2t)
  gemm_wmma<half_t, half_t, half_t, true><<<gSub, 256, 0, stream>>>(
      PSw, W2t, SDw, bsp, MSUB, DMODEL, 192, 192, 192, DMODEL, 0, 0, 0, 1.f);

  gemm_wmma<half_t, half_t, half_t, true><<<gSub, 256, 0, stream>>>(
      SDw, Wsqt, QSw, bsq, MSUB, DMODEL, DMODEL, DMODEL, DMODEL, DMODEL, 0, 0, 0, 1.f);
  gemm_wmma<half_t, half_t, half_t, true><<<gSub, 256, 0, stream>>>(
      SDw, Wskt, KSw, bsk, MSUB, DMODEL, DMODEL, DMODEL, DMODEL, DMODEL, 0, 0, 0, 1.f);
  gemm_wmma<half_t, half_t, half_t, true><<<gSub, 256, 0, stream>>>(
      SDw, Wsvt, VSw, bsv, MSUB, DMODEL, DMODEL, DMODEL, DMODEL, DMODEL, 0, 0, 0, 1.f);

  // ---- spectral attention: S = qs ks^T / sqrt(512), softmax, A @ vs
  gemm_wmma<half_t, half_t, float, true><<<dim3(7, 7, 4), 256, 0, stream>>>(
      QSw, KSw, SSw, nullptr, NSUB, NSUB, DMODEL, DMODEL, DMODEL, NSUB,
      (long)NSUB * DMODEL, (long)NSUB * DMODEL, (long)NSUB * NSUB,
      0.04419417382415922f);
  softmax_rows<<<MSUB, 256, 0, stream>>>(SSw, NSUB);
  gemm_wmma<float, half_t, half_t, false><<<dim3(7, 4, 4), 256, 0, stream>>>(
      SSw, VSw, SOP, nullptr, NSUB, DMODEL, NSUB, NSUB, DMODEL, DMODEL,
      (long)NSUB * NSUB, (long)NSUB * DMODEL, (long)NSUB * DMODEL, 1.f);
  gemm_wmma<half_t, half_t, half_t, true><<<gSub, 256, 0, stream>>>(
      SOP, Wsot, SOw, bso, MSUB, DMODEL, DMODEL, DMODEL, DMODEL, DMODEL, 0, 0, 0, 1.f);

  // ---- blend + LayerNorm (nearest upsample of spectral branch inside)
  combine_ln<<<NTOK, 128, 0, stream>>>(tokens, SPw, SOw, alpha, out);

  (void)in_sizes; (void)n_in; (void)out_size; (void)ws_size;
}
